// ComplEx_KBLN_79164837200207
// MI455X (gfx1250) — compile-verified, matrix operand order checked by reading, not run on hardware
//
#include <hip/hip_runtime.h>
#include <hip/hip_bf16.h>

// ComplEx + KBLN scoring for MI455X (gfx1250, wave32, WMMA).
// score_l collapses to one GEMM q(128x200) @ E.T(200x14951) -> 7x v_wmma_f32_16x16x32_f16.
// score_n = sum_l w*exp(-((x - lit)^2)/var) dominates (222M v_exp_f32).
// Hot loop is trans-pipe bound; loads split between DS (literal tile, 8x reused,
// loaded once per chunk) and VMEM (x/w broadcast rows, L0-resident).

static constexpr int B_  = 128;
static constexpr int NE_ = 14951;
static constexpr int D_  = 200;
static constexpr int D2_ = 100;
static constexpr int NL_ = 116;
static constexpr int QP_ = 224;   // K padded to 7*32 for WMMA 16x16x32

typedef __attribute__((ext_vector_type(16))) _Float16 v16h;
typedef __attribute__((ext_vector_type(8)))  _Float16 h8;
typedef __attribute__((ext_vector_type(8)))  float    v8f;
typedef __attribute__((ext_vector_type(4)))  float    f4;

static constexpr float LOG2E = 1.442695040888963f;

// ---------------- prep: build q (f16, padded), xs = (n_h - c)*rsqrt(var), w ----------------
__global__ __launch_bounds__(128)
void kbln_prep(const int* __restrict__ e1_idx, const int* __restrict__ r_idx,
               const float* __restrict__ E, const float* __restrict__ R,
               const float* __restrict__ nfw, const float* __restrict__ lits,
               const float* __restrict__ c, const float* __restrict__ var,
               _Float16* __restrict__ qws, float* __restrict__ xs_ws,
               float* __restrict__ w_ws)
{
    const int b = blockIdx.x;
    const int t = threadIdx.x;
    const int e1 = e1_idx[b];
    const int r  = r_idx[b];
    const float* erow = E + (long)e1 * D_;
    const float* rrow = R + (long)r  * D_;

    for (int k = t; k < D2_; k += 128) {
        float er = erow[k],        ei = erow[k + D2_];
        float rr = rrow[k],        ri = rrow[k + D2_];
        qws[b * QP_ + k]        = (_Float16)(er * rr - ei * ri);  // q_real
        qws[b * QP_ + D2_ + k]  = (_Float16)(er * ri + ei * rr);  // q_img
    }
    if (t < QP_ - D_) qws[b * QP_ + D_ + t] = (_Float16)0.0f;     // zero pad K=200..223

    for (int l = t; l < NL_; l += 128) {
        float s = __builtin_amdgcn_rsqf(var[l]);
        xs_ws[b * NL_ + l] = (lits[(long)e1 * NL_ + l] - c[l]) * s;
        w_ws [b * NL_ + l] = nfw[(long)r * NL_ + l];
    }
}

// ---------------- main: 4 waves/block, each wave = one 16x16 tile of (B x NE) ----------------
__global__ __launch_bounds__(128)
void kbln_main(const float* __restrict__ E,
               const float* __restrict__ lits,
               const float* __restrict__ var,
               const _Float16* __restrict__ qws,
               const float* __restrict__ xs_ws,
               const float* __restrict__ w_ws,
               float* __restrict__ out)
{
    __shared__ float lit_s[64 * NL_];   // [col-in-tile][l], pre-scaled by rsqrt(var)

    const int tid  = threadIdx.x;
    const int wave = tid >> 5;
    const int lane = tid & 31;
    const int n    = lane & 15;   // N index (also M index for A-fragment lanes)
    const int g    = lane >> 4;   // lane-group: K striping select
    const int rowBase  = blockIdx.y << 4;
    const int colBase0 = blockIdx.x << 6;

    // ---- stage literal tile into LDS (scaled by rsqrt(var)) ----
    for (int idx = tid; idx < 64 * NL_; idx += 128) {
        int cc = idx / NL_, l = idx % NL_;
        int col = colBase0 + cc; col = (col < NE_) ? col : (NE_ - 1);
        float s = __builtin_amdgcn_rsqf(var[l]);
        lit_s[cc * NL_ + l] = lits[(long)col * NL_ + l] * s;
    }

    // ---- WMMA GEMM: score_l tile, K = 7 x 32 (last step masked on B) ----
    const int colBase = colBase0 + (wave << 4);
    int colB = colBase + n; colB = (colB < NE_) ? colB : (NE_ - 1);
    const float*    erow = E   + (long)colB * D_;
    const _Float16* qrow = qws + (rowBase + n) * QP_;

    v8f acc = {};
    #pragma unroll
    for (int it = 0; it < 6; ++it) {
        const int kb = it * 32;
        // A fragment: lane-group g holds K in {kb+8g..+7} and {kb+16+8g..+7}
        h8 alo = *(const h8*)(qrow + kb + 8 * g);
        h8 ahi = *(const h8*)(qrow + kb + 16 + 8 * g);
        v16h a;
        #pragma unroll
        for (int e = 0; e < 8; ++e) { a[e] = alo[e]; a[e + 8] = ahi[e]; }
        // B fragment: same K striping, column n = E row colB (B is column-major in memory)
        f4 f0 = *(const f4*)(erow + kb + 8 * g);
        f4 f1 = *(const f4*)(erow + kb + 8 * g + 4);
        f4 f2 = *(const f4*)(erow + kb + 16 + 8 * g);
        f4 f3 = *(const f4*)(erow + kb + 16 + 8 * g + 4);
        v16h b;
        #pragma unroll
        for (int e = 0; e < 4; ++e) {
            b[e]      = (_Float16)f0[e];
            b[e + 4]  = (_Float16)f1[e];
            b[e + 8]  = (_Float16)f2[e];
            b[e + 12] = (_Float16)f3[e];
        }
        acc = __builtin_amdgcn_wmma_f32_16x16x32_f16(
            false, a, false, b, (short)0, acc, false, false);
    }
    {   // tail K-step kb=192: only lane-group 0 run1 (K=192..199) is in-bounds;
        // A is zero-padded past K=199 so masked-out B lanes contribute 0.
        h8 alo = *(const h8*)(qrow + 192 + 8 * g);
        h8 ahi = *(const h8*)(qrow + 208 + 8 * g);
        v16h a;
        #pragma unroll
        for (int e = 0; e < 8; ++e) { a[e] = alo[e]; a[e + 8] = ahi[e]; }
        v16h b;
        #pragma unroll
        for (int e = 0; e < 16; ++e) b[e] = (_Float16)0.0f;
        if (g == 0) {
            f4 f0 = *(const f4*)(erow + 192);
            f4 f1 = *(const f4*)(erow + 196);
            #pragma unroll
            for (int e = 0; e < 4; ++e) { b[e] = (_Float16)f0[e]; b[e + 4] = (_Float16)f1[e]; }
        }
        acc = __builtin_amdgcn_wmma_f32_16x16x32_f16(
            false, a, false, b, (short)0, acc, false, false);
    }

    __syncthreads();   // literal staging visible before RBF loop

    // ---- score_n (dominant): outer l (rolled), inner 8 rows (unrolled).
    // lit vector: 1 ds_load_b128 per 4-l chunk, reused by all 8 rows.
    // x/w rows: broadcast global_load_b128 (2 distinct addrs/wave), L0-resident.
    const float* litcol = lit_s  + ((wave << 4) + n) * NL_;
    const float* xbase  = xs_ws  + (long)(rowBase + 8 * g) * NL_;   // rows m = 8g..8g+7
    const float* wbase  = w_ws   + (long)(rowBase + 8 * g) * NL_;

    float sn[8];
    #pragma unroll
    for (int j = 0; j < 8; ++j) sn[j] = 0.0f;

    for (int l = 0; l < NL_; l += 4) {
        f4 lv = *(const f4*)(litcol + l);
        #pragma unroll
        for (int j = 0; j < 8; ++j) {
            f4 xv = *(const f4*)(xbase + j * NL_ + l);
            f4 wv = *(const f4*)(wbase + j * NL_ + l);
            #pragma unroll
            for (int u = 0; u < 4; ++u) {
                float t = xv[u] - lv[u];
                float e = __builtin_amdgcn_exp2f(t * t * (-LOG2E));
                sn[j] = fmaf(wv[u], e, sn[j]);
            }
        }
    }

    const int col = colBase + n;
    #pragma unroll
    for (int j = 0; j < 8; ++j) {
        const int m = j + 8 * g;            // C/D layout: VGPR j -> M = j + 8*g
        float x  = acc[j] + sn[j];
        float sg = __builtin_amdgcn_rcpf(1.0f + __builtin_amdgcn_exp2f(x * (-LOG2E)));
        if (col < NE_) out[(long)(rowBase + m) * NE_ + col] = sg;
    }
}

extern "C" void kernel_launch(void* const* d_in, const int* in_sizes, int n_in,
                              void* d_out, int out_size, void* d_ws, size_t ws_size,
                              hipStream_t stream) {
    const int*   e1_idx = (const int*)d_in[0];
    const int*   r_idx  = (const int*)d_in[1];
    const float* E      = (const float*)d_in[2];
    const float* R      = (const float*)d_in[3];
    const float* nfw    = (const float*)d_in[4];
    const float* lits   = (const float*)d_in[5];
    const float* c      = (const float*)d_in[6];
    const float* var    = (const float*)d_in[7];
    float*       out    = (float*)d_out;

    char* ws = (char*)d_ws;
    _Float16* qws   = (_Float16*)ws;                          // 128*224*2 = 57344 B
    float*    xs_ws = (float*)(ws + 57344);                   // 128*116*4 = 59392 B
    float*    w_ws  = (float*)(ws + 57344 + 59392);           // 128*116*4 = 59392 B

    kbln_prep<<<dim3(B_), dim3(128), 0, stream>>>(
        e1_idx, r_idx, E, R, nfw, lits, c, var, qws, xs_ws, w_ws);

    dim3 grid((NE_ + 63) / 64, B_ / 16);   // 234 x 8 blocks, 4 waves each
    kbln_main<<<grid, dim3(128), 0, stream>>>(E, lits, var, qws, xs_ws, w_ws, out);
}
